// AttentionRNN_85633057948155
// MI455X (gfx1250) — matrix operand order repaired and checked by reading, not compiled, not run on hardware
//
#include <hip/hip_runtime.h>
#include <hip/hip_bf16.h>

// Problem constants (match reference)
#define HD   2048          // hidden size
#define LL   512           // sequence length / recurrent steps
#define GG   8192          // 4*HD gate rows
#define KK   4096          // concatenated K dim: [x ; h]
#define NBLK 64            // persistent workgroups; each owns 32 hidden indices
#define NTHR 256

typedef __attribute__((ext_vector_type(16))) __bf16 v16bf;
typedef __attribute__((ext_vector_type(8)))  float  v8f;

__device__ __forceinline__ float sigf(float x) { return 1.0f / (1.0f + __expf(-x)); }

// Grid-wide sense-reversal barrier: bar[0]=count, bar[1]=generation
__device__ __forceinline__ void grid_sync(unsigned* bar) {
  __threadfence();
  __syncthreads();
  if (threadIdx.x == 0) {
    unsigned gen = __atomic_load_n(bar + 1, __ATOMIC_RELAXED);
    unsigned arr = __atomic_fetch_add(bar, 1u, __ATOMIC_ACQ_REL);
    if (arr == (unsigned)(NBLK - 1)) {
      __atomic_store_n(bar, 0u, __ATOMIC_RELAXED);
      __atomic_fetch_add(bar + 1, 1u, __ATOMIC_RELEASE);
    } else {
      while (__atomic_load_n(bar + 1, __ATOMIC_ACQUIRE) == gen)
        __builtin_amdgcn_s_sleep(1);
    }
  }
  __syncthreads();
}

// ---------------------------------------------------------------------------
// One-time kernels
// ---------------------------------------------------------------------------
__global__ void k_init(float* c, __bf16* vcat, float* wa, unsigned* bar) {
  int j = blockIdx.x * blockDim.x + threadIdx.x;  // 0..4095
  if (j < HD) c[j] = 0.0f;
  if (j < KK) vcat[j] = (__bf16)0.0f;
  if (j < 2)  wa[j] = 0.0f;
  if (j < 64) bar[j] = 0u;
}

__global__ void k_bias(const float* __restrict__ bih, const float* __restrict__ bhh,
                       float* __restrict__ bsum) {
  int m = blockIdx.x * blockDim.x + threadIdx.x;
  if (m < GG) bsum[m] = bih[m] + bhh[m];
}

// u_a[l] = input[l,:] . conv_w + conv_b  (one block per position)
__global__ void k_ua(const float* __restrict__ input, const float* __restrict__ convw,
                     const float* __restrict__ convb, float* __restrict__ ua) {
  __shared__ float s[NTHR];
  const int l = blockIdx.x, tid = threadIdx.x;
  float p = 0.0f;
  for (int i = tid; i < HD; i += NTHR) p += input[(size_t)l * HD + i] * convw[i];
  s[tid] = p; __syncthreads();
  for (int st = NTHR / 2; st > 0; st >>= 1) { if (tid < st) s[tid] += s[tid + st]; __syncthreads(); }
  if (tid == 0) ua[l] = s[0] + convb[0];
}

// Convert [W_ih | W_hh] (f32, row-major [8192 x 2048] each) into bf16 in the
// per-lane WMMA 16x16x32 A layout, k = 16*(lane/16)+e (B matches): blob[tile][kchunk][lane][16].
// Each lane then loads 32 contiguous bytes per WMMA iteration.
__global__ void k_convert(const float* __restrict__ wih, const float* __restrict__ whh,
                          __bf16* __restrict__ wcat) {
  int gidx = blockIdx.x * blockDim.x + threadIdx.x;   // (T, kc, lane)
  int lane = gidx & 31;
  int kc   = (gidx >> 5) & 127;
  int T    = gidx >> 12;
  if (T >= GG / 16) return;
  int m    = T * 16 + (lane & 15);
  int koff = (lane >= 16) ? 8 : 0;
  __bf16* dst = wcat + ((size_t)((T * 128 + kc) * 32 + lane)) * 16;
  int kbase = kc * 32;
#pragma unroll
  for (int e = 0; e < 16; ++e) {
    int k = kbase + 2 * koff + e;
    float v = (k < HD) ? wih[(size_t)m * HD + k] : whh[(size_t)m * HD + (k - HD)];
    dst[e] = (__bf16)v;
  }
}

// ---------------------------------------------------------------------------
// Persistent recurrent kernel: 512 steps, 2 grid barriers per step.
// Block b owns hidden indices j in [b*32, b*32+32). Its 8 waves compute the
// 128 gate rows {j, j+2048, j+4096, j+6144} it needs, into LDS, so the LSTM
// pointwise update fuses behind a __syncthreads().
// ---------------------------------------------------------------------------
__global__ void __launch_bounds__(NTHR)
k_recurrent(const float* __restrict__ input, const float* __restrict__ bias_mat,
            const float* __restrict__ fc1w, const float* __restrict__ fc1b,
            const __bf16* __restrict__ wcat, const float* __restrict__ bsum,
            const float* __restrict__ ua,
            float* __restrict__ cS, __bf16* __restrict__ vcat,
            float* __restrict__ wa, unsigned* __restrict__ bar,
            float* __restrict__ out) {
  const int tid  = threadIdx.x;
  const int blk  = blockIdx.x;
  const int lane = tid & 31;
  const int wave = tid >> 5;
  const int sec  = wave >> 1;                 // gate section: 0=i,1=f,2=g,3=o
  const int half = wave & 1;                  // which 16 of the 32 owned rows
  const int T    = sec * 128 + blk * 2 + half;// tile id in swizzled weight layout
  const int bsel = (lane >= 16) ? 1 : 0;

  __shared__ float sred[NTHR];
  __shared__ float sattn[LL];
  __shared__ float sctx[8][32];
  __shared__ float sgates[4][32];

  const v16bf* Ab = (const v16bf*)wcat + (size_t)T * 128 * 32;
  const v16bf* Bb = (const v16bf*)vcat;

  for (int t = 0; t < LL; ++t) {
    // ---- Phase A: redundant softmax (per block) + context slice ------------
    float wav = wa[t & 1] + fc1b[0];
    if (blk == 0 && tid == 0) wa[(t + 1) & 1] = 0.0f;  // reset next-step accumulator

    int l0 = tid * 2, l1 = l0 + 1;
    float z0 = ua[l0] + wav; z0 = (z0 > 0.0f) ? z0 : 0.01f * z0; z0 += bias_mat[l0];
    float z1 = ua[l1] + wav; z1 = (z1 > 0.0f) ? z1 : 0.01f * z1; z1 += bias_mat[l1];

    sred[tid] = fmaxf(z0, z1); __syncthreads();
    for (int s = NTHR / 2; s > 0; s >>= 1) { if (tid < s) sred[tid] = fmaxf(sred[tid], sred[tid + s]); __syncthreads(); }
    float mx = sred[0]; __syncthreads();

    float e0 = __expf(z0 - mx), e1 = __expf(z1 - mx);
    sred[tid] = e0 + e1; __syncthreads();
    for (int s = NTHR / 2; s > 0; s >>= 1) { if (tid < s) sred[tid] += sred[tid + s]; __syncthreads(); }
    float inv = 1.0f / sred[0];
    sattn[l0] = e0 * inv; sattn[l1] = e1 * inv;
    __syncthreads();

    // context: this block computes x[blk*32 .. blk*32+31]
    {
      int col   = blk * 32 + lane;            // lane == tid & 31
      int slice = wave;                       // 8 slices of 64 positions
      float p = 0.0f;
      int lbeg = slice * 64;
      for (int l = lbeg; l < lbeg + 64; ++l) p += sattn[l] * input[(size_t)l * HD + col];
      sctx[slice][lane] = p;
      __syncthreads();
      if (tid < 32) {
        float x = 0.0f;
#pragma unroll
        for (int r = 0; r < 8; ++r) x += sctx[r][tid];
        vcat[blk * 32 + tid] = (__bf16)x;
      }
    }
    grid_sync(bar);

    // ---- Phase B+C fused: gates via WMMA -> LDS -> LSTM pointwise ----------
    {
      v8f acc0 = {0.f, 0.f, 0.f, 0.f, 0.f, 0.f, 0.f, 0.f};
      v8f acc1 = {0.f, 0.f, 0.f, 0.f, 0.f, 0.f, 0.f, 0.f};
      for (int kc = 0; kc < 128; kc += 2) {
        v16bf a0 = Ab[kc * 32 + lane];
        v16bf b0 = Bb[kc * 2 + bsel];
        v16bf a1 = Ab[(kc + 1) * 32 + lane];
        v16bf b1 = Bb[(kc + 1) * 2 + bsel];
        acc0 = __builtin_amdgcn_wmma_f32_16x16x32_bf16(false, a0, false, b0, (short)0, acc0, false, false);
        acc1 = __builtin_amdgcn_wmma_f32_16x16x32_bf16(false, a1, false, b1, (short)0, acc1, false, false);
      }
      if ((lane & 15) == 0) {                 // lane 0 -> rows 0..7, lane 16 -> rows 8..15
        int rl    = half * 16 + ((lane >= 16) ? 8 : 0);  // local row in [0,32)
        int rbase = sec * 2048 + blk * 32 + rl;          // global gate row
#pragma unroll
        for (int r = 0; r < 8; ++r)
          sgates[sec][rl + r] = acc0[r] + acc1[r] + bsum[rbase + r];
      }
    }
    __syncthreads();

    if (tid < 32) {                           // pointwise for this block's 32 j's
      int j = blk * 32 + tid;
      float ig = sgates[0][tid], fg = sgates[1][tid];
      float gg = sgates[2][tid], og = sgates[3][tid];
      float cn = sigf(fg) * cS[j] + sigf(ig) * tanhf(gg);
      float hn = sigf(og) * tanhf(cn);
      cS[j] = cn;
      out[(size_t)t * HD + j] = hn;
      vcat[HD + j] = (__bf16)hn;              // h half of next step's GEMV vector
      sred[tid] = hn * fc1w[j];               // partial of next step's w_a
    }
    __syncthreads();
    if (tid == 0) {
      float p = 0.0f;
#pragma unroll
      for (int i = 0; i < 32; ++i) p += sred[i];
      atomicAdd(&wa[(t + 1) & 1], p);
    }
    grid_sync(bar);
  }
}

// ---------------------------------------------------------------------------
extern "C" void kernel_launch(void* const* d_in, const int* in_sizes, int n_in,
                              void* d_out, int out_size, void* d_ws, size_t ws_size,
                              hipStream_t stream) {
  const float* input    = (const float*)d_in[0];   // [1, L, H]
  const float* bias_mat = (const float*)d_in[1];   // [1, 1, L]
  const float* conv_w   = (const float*)d_in[2];   // [H]
  const float* conv_b   = (const float*)d_in[3];   // [1]
  const float* fc1_w    = (const float*)d_in[4];   // [1, H]
  const float* fc1_b    = (const float*)d_in[5];   // [1]
  const float* w_ih     = (const float*)d_in[6];   // [4H, H]
  const float* b_ih     = (const float*)d_in[7];   // [4H]
  const float* w_hh     = (const float*)d_in[8];   // [4H, H]
  const float* b_hh     = (const float*)d_in[9];   // [4H]
  float* out = (float*)d_out;                      // [L, 1, H]

  // Workspace carve-up (256B aligned slices)
  size_t off = 0;
  auto carve = [&](size_t bytes) { size_t o = off; off += (bytes + 255) & ~(size_t)255; return o; };
  char* ws = (char*)d_ws;
  __bf16*   wcat = (__bf16*)  (ws + carve((size_t)GG * KK * sizeof(__bf16))); // 64 MB swizzled bf16 weights
  float*    bsum = (float*)   (ws + carve((size_t)GG * sizeof(float)));
  float*    ua   = (float*)   (ws + carve((size_t)LL * sizeof(float)));
  float*    c    = (float*)   (ws + carve((size_t)HD * sizeof(float)));
  __bf16*   vcat = (__bf16*)  (ws + carve((size_t)KK * sizeof(__bf16)));
  float*    wa   = (float*)   (ws + carve(2 * sizeof(float)));
  unsigned* bar  = (unsigned*)(ws + carve(256));
  (void)ws_size; (void)in_sizes; (void)n_in; (void)out_size;

  k_init   <<<16,   NTHR, 0, stream>>>(c, vcat, wa, bar);
  k_bias   <<<GG / NTHR, NTHR, 0, stream>>>(b_ih, b_hh, bsum);
  k_ua     <<<LL,   NTHR, 0, stream>>>(input, conv_w, conv_b, ua);
  k_convert<<<8192, NTHR, 0, stream>>>(w_ih, w_hh, wcat);
  k_recurrent<<<NBLK, NTHR, 0, stream>>>(input, bias_mat, fc1_w, fc1_b, wcat, bsum, ua,
                                         c, vcat, wa, bar, out);
}
